// SlidingWindowAttention_65687229825449
// MI455X (gfx1250) — matrix operand order repaired and checked by reading, not compiled, not run on hardware
//
#include <hip/hip_runtime.h>
#include <hip/hip_bf16.h>

typedef __attribute__((ext_vector_type(16))) __bf16 v16bf;
typedef __attribute__((ext_vector_type(8)))  __bf16 v8bf;
typedef __attribute__((ext_vector_type(8)))  float  v8f;

#define S_LEN 4096
#define D_DIM 64
#define H_NUM 16
#define WIN   512
#define QTILE 64
#define KTILE 64
#define KPAD  72   // Ks/Vt padded row length (halves): 144B rows, 16B aligned
#define PPAD  72   // P staging padded row length (halves): 144B rows, 16B aligned
#define VROWS 80   // 64 d-rows + 16 extra rows (row 64 = ones column for row-sum WMMA)

__device__ __forceinline__ float hmax16(float x) {
    x = fmaxf(x, __shfl_xor(x, 1));
    x = fmaxf(x, __shfl_xor(x, 2));
    x = fmaxf(x, __shfl_xor(x, 4));
    x = fmaxf(x, __shfl_xor(x, 8));
    return x;
}

__device__ __forceinline__ unsigned pack_bf2(float a, float b) {
    unsigned short ua = __builtin_bit_cast(unsigned short, (__bf16)a);
    unsigned short ub = __builtin_bit_cast(unsigned short, (__bf16)b);
    return (unsigned)ua | ((unsigned)ub << 16);
}

__global__ __launch_bounds__(128)
void swa_wmma_kernel(const float* __restrict__ Q, const float* __restrict__ K,
                     const float* __restrict__ V, float* __restrict__ O) {
    __shared__ __bf16 Ks[KTILE * KPAD];     // [key][d]
    __shared__ __bf16 Vt[VROWS * KPAD];     // [d][key] transposed; rows 64..79: [1;0...]
    __shared__ __bf16 Ps[4 * 16 * PPAD];    // per-wave P staging (C->A relayout)

    const int qb   = blockIdx.x;
    const int bh   = blockIdx.z * H_NUM + blockIdx.y;
    const int tid  = threadIdx.x;
    const int wave = tid >> 5;
    const int lane = tid & 31;
    const int hi   = lane >> 4;
    const int ln   = lane & 15;
    const int c0   = hi * 8;                // A-layout K sub-chunk offset

    const int qs     = qb * QTILE;
    const int qbase  = qs + wave * 16;
    const bool causal = (qb != 0);          // block 0 == 64 global queries (all keys, no mask)

    const size_t base = (size_t)bh * S_LEN * D_DIM;
    const float* q = Q + base;
    const float* k = K + base;
    const float* v = V + base;
    float*       o = O + base;

    // ---- ones/zeros rows for the row-sum column (init once; covered by loop-top barrier) ----
    for (int it = tid; it < 16 * KTILE; it += 128) {
        const int r   = it >> 6;
        const int key = it & 63;
        Vt[(64 + r) * KPAD + key] = (r == 0) ? (__bf16)1.0f : (__bf16)0.0f;
    }

    // ---- Q fragments (A layout), fp32 -> bf16 ----
    v16bf a0, a1;
    {
        const float* qr = q + (size_t)(qbase + ln) * D_DIM;
        const float4 f0 = *(const float4*)(qr + c0);
        const float4 f1 = *(const float4*)(qr + c0 + 4);
        const float4 f2 = *(const float4*)(qr + 16 + c0);
        const float4 f3 = *(const float4*)(qr + 16 + c0 + 4);
        const float4 f4 = *(const float4*)(qr + 32 + c0);
        const float4 f5 = *(const float4*)(qr + 32 + c0 + 4);
        const float4 f6 = *(const float4*)(qr + 48 + c0);
        const float4 f7 = *(const float4*)(qr + 48 + c0 + 4);
        a0[0]=(__bf16)f0.x; a0[1]=(__bf16)f0.y; a0[2]=(__bf16)f0.z; a0[3]=(__bf16)f0.w;
        a0[4]=(__bf16)f1.x; a0[5]=(__bf16)f1.y; a0[6]=(__bf16)f1.z; a0[7]=(__bf16)f1.w;
        a0[8]=(__bf16)f2.x; a0[9]=(__bf16)f2.y; a0[10]=(__bf16)f2.z; a0[11]=(__bf16)f2.w;
        a0[12]=(__bf16)f3.x; a0[13]=(__bf16)f3.y; a0[14]=(__bf16)f3.z; a0[15]=(__bf16)f3.w;
        a1[0]=(__bf16)f4.x; a1[1]=(__bf16)f4.y; a1[2]=(__bf16)f4.z; a1[3]=(__bf16)f4.w;
        a1[4]=(__bf16)f5.x; a1[5]=(__bf16)f5.y; a1[6]=(__bf16)f5.z; a1[7]=(__bf16)f5.w;
        a1[8]=(__bf16)f6.x; a1[9]=(__bf16)f6.y; a1[10]=(__bf16)f6.z; a1[11]=(__bf16)f6.w;
        a1[12]=(__bf16)f7.x; a1[13]=(__bf16)f7.y; a1[14]=(__bf16)f7.z; a1[15]=(__bf16)f7.w;
    }

    v8f acc[5] = {};                        // 4 output d-tiles + row-sum column tile
    float mrow[8];
    #pragma unroll
    for (int j = 0; j < 8; ++j) mrow[j] = -3.0e38f;

    const float scale2 = 0.18033688011112042f;   // log2(e)/sqrt(D)

    int kstart = 0, kend = S_LEN;
    if (causal) {
        kstart = qs + QTILE - WIN - KTILE;       // tile-aligned
        if (kstart < 0) kstart = 0;
        kend = qs + QTILE;
    }

    __bf16* psme = Ps + wave * 16 * PPAD;

    for (int kt = kstart; kt < kend; kt += KTILE) {
        __syncthreads();
        // ---- cooperative stage (no bounds checks needed: key range always in [0,S)) ----
        #pragma unroll 2
        for (int it = tid; it < KTILE * (D_DIM / 4); it += 128) {
            const int key = it >> 4;
            const int d4  = (it & 15) * 4;
            const float4 kv4 = *(const float4*)(k + (size_t)(kt + key) * D_DIM + d4);
            const float4 vv4 = *(const float4*)(v + (size_t)(kt + key) * D_DIM + d4);
            uint2 pk;
            pk.x = pack_bf2(kv4.x, kv4.y);
            pk.y = pack_bf2(kv4.z, kv4.w);
            *(uint2*)(Ks + key * KPAD + d4) = pk;      // one b64 store
            Vt[(d4 + 0) * KPAD + key] = (__bf16)vv4.x;
            Vt[(d4 + 1) * KPAD + key] = (__bf16)vv4.y;
            Vt[(d4 + 2) * KPAD + key] = (__bf16)vv4.z;
            Vt[(d4 + 3) * KPAD + key] = (__bf16)vv4.w;
        }
        __syncthreads();

        // ---- scores for the whole 64-key tile: 4 fragments of 16 keys ----
        v8f sfr[4];
        #pragma unroll
        for (int nt = 0; nt < 4; ++nt) {
            v8f c = {};
            #pragma unroll
            for (int kc = 0; kc < 2; ++kc) {
                const __bf16* bp = Ks + (nt * 16 + ln) * KPAD + kc * 32 + hi * 8;
                v8bf b0 = *(const v8bf*)bp;
                v8bf b1 = *(const v8bf*)(bp + 16);
                v16bf bb;
                #pragma unroll
                for (int t = 0; t < 8; ++t) { bb[t] = b0[t]; bb[8 + t] = b1[t]; }
                c = __builtin_amdgcn_wmma_f32_16x16x32_bf16(
                        false, (kc ? a1 : a0), false, bb, (short)0, c, false, false);
            }
            const int kg0 = kt + nt * 16;
            // uniform: fragment fully inside the window for every row of this wave?
            const bool needmask = causal && !((kg0 >= qbase - (WIN - 16)) && (kg0 + 15 <= qbase));
            if (needmask) {
                const int d0 = qbase + 8 * hi - (kg0 + ln);   // i - key for j = 0
                #pragma unroll
                for (int j = 0; j < 8; ++j)
                    c[j] = ((unsigned)(d0 + j) < WIN) ? c[j] * scale2 : -1.0e30f;
            } else {
                #pragma unroll
                for (int j = 0; j < 8; ++j) c[j] *= scale2;
            }
            sfr[nt] = c;
        }

        // ---- online softmax over 64 keys (row-sum handled by the ones-column WMMA) ----
        #pragma unroll
        for (int j = 0; j < 8; ++j) {
            float mx = fmaxf(fmaxf(sfr[0][j], sfr[1][j]), fmaxf(sfr[2][j], sfr[3][j]));
            mx = hmax16(mx);
            const float mn    = fmaxf(mrow[j], mx);
            const float alpha = __builtin_amdgcn_exp2f(mrow[j] - mn);
            mrow[j] = mn;
            #pragma unroll
            for (int f = 0; f < 5; ++f) acc[f][j] *= alpha;
            #pragma unroll
            for (int nt = 0; nt < 4; ++nt)
                sfr[nt][j] = __builtin_amdgcn_exp2f(sfr[nt][j] - mn);
        }

        // ---- relayout P (C-layout -> A-layout) through per-wave LDS ----
        #pragma unroll
        for (int nt = 0; nt < 4; ++nt)
            #pragma unroll
            for (int j = 0; j < 8; ++j)
                psme[(j + 8 * hi) * PPAD + nt * 16 + ln] = (__bf16)sfr[nt][j];
        v16bf pa0, pa1;
        {
            const __bf16* pr = psme + ln * PPAD;
            v8bf p0 = *(const v8bf*)(pr + c0);
            v8bf p1 = *(const v8bf*)(pr + 16 + c0);
            v8bf p2 = *(const v8bf*)(pr + 32 + c0);
            v8bf p3 = *(const v8bf*)(pr + 48 + c0);
            #pragma unroll
            for (int t = 0; t < 8; ++t) {
                pa0[t] = p0[t]; pa0[8 + t] = p1[t];
                pa1[t] = p2[t]; pa1[8 + t] = p3[t];
            }
        }

        // ---- O += P x [V | 1] : 5 d-tiles x 2 key-chunks of 32 ----
        #pragma unroll
        for (int f = 0; f < 5; ++f) {
            #pragma unroll
            for (int kc = 0; kc < 2; ++kc) {
                const __bf16* vp = Vt + (f * 16 + ln) * KPAD + kc * 32 + hi * 8;
                v8bf b0 = *(const v8bf*)vp;
                v8bf b1 = *(const v8bf*)(vp + 16);
                v16bf bb;
                #pragma unroll
                for (int t = 0; t < 8; ++t) { bb[t] = b0[t]; bb[8 + t] = b1[t]; }
                acc[f] = __builtin_amdgcn_wmma_f32_16x16x32_bf16(
                             false, (kc ? pa1 : pa0), false, bb, (short)0, acc[f], false, false);
            }
        }
    }

    // ---- epilogue: l lives in acc[4] column 0 (lanes ln==0 of each half); broadcast, normalize ----
    #pragma unroll
    for (int j = 0; j < 8; ++j) {
        const float lv   = __shfl(acc[4][j], lane & 16);
        const float invl = __builtin_amdgcn_rcpf(lv);
        const int i = qbase + 8 * hi + j;
        #pragma unroll
        for (int f = 0; f < 4; ++f)
            o[(size_t)i * D_DIM + f * 16 + ln] = acc[f][j] * invl;
    }
}

extern "C" void kernel_launch(void* const* d_in, const int* in_sizes, int n_in,
                              void* d_out, int out_size, void* d_ws, size_t ws_size,
                              hipStream_t stream) {
    (void)n_in; (void)out_size; (void)d_ws; (void)ws_size;
    const float* q = (const float*)d_in[0];
    const float* k = (const float*)d_in[1];
    const float* v = (const float*)d_in[2];
    float* out = (float*)d_out;
    const int B = in_sizes[0] / (H_NUM * S_LEN * D_DIM);
    dim3 grid(S_LEN / QTILE, H_NUM, B);
    dim3 block(128);
    swa_wmma_kernel<<<grid, block, 0, stream>>>(q, k, v, out);
}
